// NetworkAction_74225624809874
// MI455X (gfx1250) — matrix operand order repaired and checked by reading, not compiled, not run on hardware
//
#include <hip/hip_runtime.h>
#include <hip/hip_bf16.h>
#include <math.h>

typedef __attribute__((ext_vector_type(16))) _Float16 v16h;
typedef __attribute__((ext_vector_type(8)))  float    v8f;

#define TOPK   16
#define NPTS   4096
#define NBATCH 4
#define NODE_STRIDE 160   // 132 real channels padded with zeros to 5 k-chunks of 32

// ---------------------------------------------------------------------------
// Kernel 1: exact-KNN (top-16 smallest d2, ties -> lowest index, matches
// jax.lax.top_k(-d2)). One thread per point, candidate tiles staged in LDS.
// ---------------------------------------------------------------------------
__global__ __launch_bounds__(256) void knn_kernel(const float* __restrict__ s,
                                                  int* __restrict__ knn_out) {
    const int b   = blockIdx.x >> 4;                       // 16 blocks per batch
    const int i   = ((blockIdx.x & 15) << 8) + threadIdx.x; // 0..4095
    const float* sb = s + (size_t)b * NPTS * 4;

    const float xi  = sb[i * 4 + 0];
    const float yi  = sb[i * 4 + 1];
    const float sqi = xi * xi + yi * yi;

    float bd[TOPK];
    int   bi[TOPK];
#pragma unroll
    for (int k = 0; k < TOPK; ++k) { bd[k] = 3.4e38f; bi[k] = 0; }

    __shared__ float tx[256], ty[256], tq[256];

    for (int t = 0; t < NPTS / 256; ++t) {
        const int m0 = t * 256;
        __syncthreads();
        {
            const float x = sb[(m0 + threadIdx.x) * 4 + 0];
            const float y = sb[(m0 + threadIdx.x) * 4 + 1];
            tx[threadIdx.x] = x; ty[threadIdx.x] = y; tq[threadIdx.x] = x * x + y * y;
        }
        __syncthreads();

        for (int j = 0; j < 256; ++j) {
            const float d = sqi + tq[j] - 2.0f * (xi * tx[j] + yi * ty[j]);
            const int   m = m0 + j;
            if (d < bd[TOPK - 1]) {
                // branchless-ish stable insertion (strict <, equal keys keep
                // earlier index first -> matches top_k tie-breaking)
#pragma unroll
                for (int k = TOPK - 1; k >= 1; --k) {
                    if (d < bd[k - 1])      { bd[k] = bd[k - 1]; bi[k] = bi[k - 1]; }
                    else if (d < bd[k])     { bd[k] = d;         bi[k] = m; }
                }
                if (d < bd[0]) { bd[0] = d; bi[0] = m; }
            }
        }
    }

    int* dst = knn_out + ((size_t)b * NPTS + i) * TOPK;
#pragma unroll
    for (int k = 0; k < TOPK; ++k) dst[k] = bi[k];
}

// ---------------------------------------------------------------------------
// Kernel 2: f32 -> f16 weight conversion
// ---------------------------------------------------------------------------
__global__ void cvt_f16_kernel(const float* __restrict__ src,
                               _Float16* __restrict__ dst, int n) {
    const int i = blockIdx.x * 256 + threadIdx.x;
    if (i < n) dst[i] = (_Float16)src[i];
}

// ---------------------------------------------------------------------------
// WMMA fragment loaders (ISA 16-bit A 16x32 layout / B 32x16 layout).
// ---------------------------------------------------------------------------
__device__ __forceinline__ v16h load_a_frag(const _Float16* mat, int stride,
                                            int row, int half, int kb) {
    v16h a;
#pragma unroll
    for (int el = 0; el < 16; ++el) {
        const int vg = el >> 1, lo = el & 1;
        int k = (vg < 4) ? (2 * vg + lo) : (16 + 2 * (vg - 4) + lo);
        k += half * 8 + kb;          // lanes 16-31 hold K shifted by 8
        a[el] = mat[row * stride + k];
    }
    return a;
}

// weights stored row-major (O x C); col = output channel, k = input channel
__device__ __forceinline__ v16h load_b_frag(const _Float16* wt, int O, int C,
                                            int col, int half, int kb) {
    v16h bf;
#pragma unroll
    for (int el = 0; el < 16; ++el) {
        const int k = kb + ((el < 8) ? el : 8 + el) + half * 8; // lanes 16-31 hold K shifted by 8
        _Float16 v = (_Float16)0.0f;
        if (col < O && k < C) v = wt[col * C + k];
        bf[el] = v;
    }
    return bf;
}

// ---------------------------------------------------------------------------
// Kernel 3: EdgeConv. One wave per point: build e(16x5), VALU layer1 -> h1
// (16x64 f16 in LDS), WMMA layer2 (16x64x128), relu+max-over-k epilogue,
// emit 132-channel node vector (f16, zero-padded to 160).
// ---------------------------------------------------------------------------
__global__ __launch_bounds__(128) void edgeconv_kernel(
    const float* __restrict__ s, const float* __restrict__ g,
    const int* __restrict__ knn,
    const float* __restrict__ w1, const float* __restrict__ b1,
    const _Float16* __restrict__ w2h, const float* __restrict__ b2,
    _Float16* __restrict__ node) {

    const int lane  = threadIdx.x & 31;
    const int wv    = threadIdx.x >> 5;
    const int point = blockIdx.x * 4 + wv;          // 0..16383
    const int b     = point >> 12;
    const int i     = point & (NPTS - 1);
    const int row   = lane & 15;
    const int half  = lane >> 4;

    __shared__ float    e_s[4][16][6];
    __shared__ _Float16 h1_s[4][16][64];

    if (lane < 16) {
        const int nb = knn[(size_t)point * TOPK + lane];
        const float* sp = s + ((size_t)b * NPTS + i) * 4;
        const float* sn = s + ((size_t)b * NPTS + nb) * 4;
#pragma unroll
        for (int c = 0; c < 4; ++c) e_s[wv][lane][c] = sp[c] - sn[c];
        e_s[wv][lane][4] = (nb == i) ? 1.0f : 0.0f;
    }
    __syncthreads();

    // layer 1: 5 -> 64, relu, store f16 (each lane: its row, 32 of 64 cols)
    for (int co = half; co < 64; co += 2) {
        float acc = b1[co];
#pragma unroll
        for (int c = 0; c < 5; ++c) acc = fmaf(e_s[wv][row][c], w1[co * 5 + c], acc);
        h1_s[wv][row][co] = (_Float16)fmaxf(acc, 0.0f);
    }
    __syncthreads();

    const v16h a0 = load_a_frag(&h1_s[wv][0][0], 64, row, half, 0);
    const v16h a1 = load_a_frag(&h1_s[wv][0][0], 64, row, half, 32);

    // layer 2: 16x64x128 GEMM via WMMA, fused bias+relu+max-over-k
#pragma unroll
    for (int nt = 0; nt < 8; ++nt) {
        const int o = nt * 16 + row;
        const v16h bb0 = load_b_frag(w2h, 128, 64, o, half, 0);
        const v16h bb1 = load_b_frag(w2h, 128, 64, o, half, 32);
        v8f c = {};
        c = __builtin_amdgcn_wmma_f32_16x16x32_f16(false, a0, false, bb0,
                                                   (short)0, c, false, false);
        c = __builtin_amdgcn_wmma_f32_16x16x32_f16(false, a1, false, bb1,
                                                   (short)0, c, false, false);
        const float bias = b2[o];
        float m = 0.0f;                 // relu floor: max over k of relu >= 0
#pragma unroll
        for (int r = 0; r < 8; ++r) m = fmaxf(m, c[r] + bias);
        m = fmaxf(m, __shfl_xor(m, 16, 32));   // merge rows 0-7 with 8-15
        if (half == 0) node[(size_t)point * NODE_STRIDE + o] = (_Float16)m;
    }

    if (lane == 0) {
        const float* sp = s + ((size_t)b * NPTS + i) * 4;
        const float* gp = g + ((size_t)b * NPTS + i) * 2;
        _Float16* np = node + (size_t)point * NODE_STRIDE;
        np[128] = (_Float16)(sp[0] - gp[0]);
        np[129] = (_Float16)(sp[1] - gp[1]);
        np[130] = (_Float16)sp[2];
        np[131] = (_Float16)sp[3];
#pragma unroll
        for (int c = 132; c < NODE_STRIDE; ++c) np[c] = (_Float16)0.0f;
    }
}

// ---------------------------------------------------------------------------
// Kernel 4: node MLP 132->64->128->64->4, 16 points per wave (M=16 WMMA).
// ---------------------------------------------------------------------------
__global__ __launch_bounds__(128) void nodemlp_kernel(
    const _Float16* __restrict__ node,
    const _Float16* __restrict__ fw1h, const float* __restrict__ fb1,
    const _Float16* __restrict__ fw2h, const float* __restrict__ fb2,
    const _Float16* __restrict__ fw3h, const float* __restrict__ fb3,
    const _Float16* __restrict__ fw4h, const float* __restrict__ fb4,
    float* __restrict__ out) {

    const int lane = threadIdx.x & 31;
    const int wv   = threadIdx.x >> 5;
    const int grp  = blockIdx.x * 4 + wv;   // 0..1023
    const int base = grp * 16;              // first point of this wave
    const int row  = lane & 15;
    const int half = lane >> 4;

    __shared__ _Float16 hA[4][16][128];
    __shared__ _Float16 hB[4][16][128];

    const _Float16* A0 = node + (size_t)base * NODE_STRIDE;

    // L1: K=160 (zero-padded 132), O=64
#pragma unroll
    for (int nt = 0; nt < 4; ++nt) {
        v8f c = {};
#pragma unroll
        for (int kc = 0; kc < 5; ++kc) {
            const v16h a  = load_a_frag(A0, NODE_STRIDE, row, half, kc * 32);
            const v16h bb = load_b_frag(fw1h, 64, 132, nt * 16 + row, half, kc * 32);
            c = __builtin_amdgcn_wmma_f32_16x16x32_f16(false, a, false, bb,
                                                       (short)0, c, false, false);
        }
        const float bias = fb1[nt * 16 + row];
#pragma unroll
        for (int r = 0; r < 8; ++r)
            hA[wv][r + half * 8][nt * 16 + row] = (_Float16)fmaxf(c[r] + bias, 0.0f);
    }
    __syncthreads();

    // L2: K=64, O=128
#pragma unroll
    for (int nt = 0; nt < 8; ++nt) {
        v8f c = {};
#pragma unroll
        for (int kc = 0; kc < 2; ++kc) {
            const v16h a  = load_a_frag(&hA[wv][0][0], 128, row, half, kc * 32);
            const v16h bb = load_b_frag(fw2h, 128, 64, nt * 16 + row, half, kc * 32);
            c = __builtin_amdgcn_wmma_f32_16x16x32_f16(false, a, false, bb,
                                                       (short)0, c, false, false);
        }
        const float bias = fb2[nt * 16 + row];
#pragma unroll
        for (int r = 0; r < 8; ++r)
            hB[wv][r + half * 8][nt * 16 + row] = (_Float16)fmaxf(c[r] + bias, 0.0f);
    }
    __syncthreads();

    // L3: K=128, O=64
#pragma unroll
    for (int nt = 0; nt < 4; ++nt) {
        v8f c = {};
#pragma unroll
        for (int kc = 0; kc < 4; ++kc) {
            const v16h a  = load_a_frag(&hB[wv][0][0], 128, row, half, kc * 32);
            const v16h bb = load_b_frag(fw3h, 64, 128, nt * 16 + row, half, kc * 32);
            c = __builtin_amdgcn_wmma_f32_16x16x32_f16(false, a, false, bb,
                                                       (short)0, c, false, false);
        }
        const float bias = fb3[nt * 16 + row];
#pragma unroll
        for (int r = 0; r < 8; ++r)
            hA[wv][r + half * 8][nt * 16 + row] = (_Float16)fmaxf(c[r] + bias, 0.0f);
    }
    __syncthreads();

    // L4: K=64, O=4 (cols 4..15 are zero via load_b guard); 2*sigmoid-1 == tanh(x/2)
    {
        v8f c = {};
#pragma unroll
        for (int kc = 0; kc < 2; ++kc) {
            const v16h a  = load_a_frag(&hA[wv][0][0], 128, row, half, kc * 32);
            const v16h bb = load_b_frag(fw4h, 4, 64, row, half, kc * 32);
            c = __builtin_amdgcn_wmma_f32_16x16x32_f16(false, a, false, bb,
                                                       (short)0, c, false, false);
        }
        if (row < 4) {
            const float bias = fb4[row];
#pragma unroll
            for (int r = 0; r < 8; ++r) {
                const int p = base + r + half * 8;
                out[(size_t)p * 4 + row] = tanhf(0.5f * (c[r] + bias));
            }
        }
    }
}

// ---------------------------------------------------------------------------
extern "C" void kernel_launch(void* const* d_in, const int* in_sizes, int n_in,
                              void* d_out, int out_size, void* d_ws, size_t ws_size,
                              hipStream_t stream) {
    (void)in_sizes; (void)n_in; (void)out_size; (void)ws_size;

    const float* s   = (const float*)d_in[0];
    const float* g   = (const float*)d_in[1];
    const float* w1  = (const float*)d_in[2];
    const float* b1  = (const float*)d_in[3];
    const float* w2  = (const float*)d_in[4];
    const float* b2  = (const float*)d_in[5];
    const float* fw1 = (const float*)d_in[6];
    const float* fb1 = (const float*)d_in[7];
    const float* fw2 = (const float*)d_in[8];
    const float* fb2 = (const float*)d_in[9];
    const float* fw3 = (const float*)d_in[10];
    const float* fb3 = (const float*)d_in[11];
    const float* fw4 = (const float*)d_in[12];
    const float* fb4 = (const float*)d_in[13];
    float* out = (float*)d_out;

    char* wsb = (char*)d_ws;
    size_t off = 0;
    auto take = [&](size_t bytes) -> void* {
        void* p = wsb + off;
        off += (bytes + 255) & ~(size_t)255;
        return p;
    };

    int*      knn   = (int*)      take((size_t)NBATCH * NPTS * TOPK * sizeof(int));
    _Float16* nodef = (_Float16*) take((size_t)NBATCH * NPTS * NODE_STRIDE * sizeof(_Float16));
    _Float16* w2h   = (_Float16*) take((size_t)128 * 64 * 2);
    _Float16* fw1h  = (_Float16*) take((size_t)64 * 132 * 2);
    _Float16* fw2h  = (_Float16*) take((size_t)128 * 64 * 2);
    _Float16* fw3h  = (_Float16*) take((size_t)64 * 128 * 2);
    _Float16* fw4h  = (_Float16*) take((size_t)4 * 64 * 2);

    knn_kernel<<<NBATCH * (NPTS / 256), 256, 0, stream>>>(s, knn);

    cvt_f16_kernel<<<(128 * 64 + 255) / 256, 256, 0, stream>>>(w2,  w2h,  128 * 64);
    cvt_f16_kernel<<<(64 * 132 + 255) / 256, 256, 0, stream>>>(fw1, fw1h, 64 * 132);
    cvt_f16_kernel<<<(128 * 64 + 255) / 256, 256, 0, stream>>>(fw2, fw2h, 128 * 64);
    cvt_f16_kernel<<<(64 * 128 + 255) / 256, 256, 0, stream>>>(fw3, fw3h, 64 * 128);
    cvt_f16_kernel<<<(4 * 64 + 255) / 256, 256, 0, stream>>>(fw4, fw4h, 4 * 64);

    // one wave per point, 4 waves per block
    edgeconv_kernel<<<(NBATCH * NPTS) / 4, 128, 0, stream>>>(
        s, g, knn, w1, b1, w2h, b2, nodef);

    // 16 points per wave, 4 waves per block
    nodemlp_kernel<<<(NBATCH * NPTS) / 16 / 4, 128, 0, stream>>>(
        nodef, fw1h, fb1, fw2h, fb2, fw3h, fb3, fw4h, fb4, out);
}
